// KMeansParamHead_90778428768744
// MI455X (gfx1250) — matrix operand order repaired and checked by reading, not compiled, not run on hardware
//
#include <hip/hip_runtime.h>
#include <stdint.h>

typedef __attribute__((ext_vector_type(16))) _Float16 v16h;
typedef __attribute__((ext_vector_type(8)))  float    v8f;
typedef __attribute__((ext_vector_type(4)))  float    v4f;
typedef __attribute__((ext_vector_type(8)))  unsigned int v8u;

#define DIM      384
#define NCLASS   256
#define KCHUNKS  12      // 384 / 32
#define CTILES   16      // 256 / 16
#define NPTS     (4*256*256)
#define EPSN     1e-12f

// B workspace: KCHUNKS * CTILES fragments, each 32 lanes * 8 dwords = 1KB
#define B_DWORDS (KCHUNKS * CTILES * 32 * 8)   // 49152 dwords = 192KB

static __device__ __forceinline__ uint32_t pack2h(float x, float y) {
    union { _Float16 h[2]; uint32_t u; } u;
    u.h[0] = (_Float16)x; u.h[1] = (_Float16)y;
    return u.u;
}

// ---------------------------------------------------------------------------
// Kernel 1: normalize cluster centers (fp32), convert to f16, store in the
// B-fragment order this file uses with V_WMMA_F32_16X16X32_F16 (contiguous k, as A):
//   B is 32x16 (KxN): lane = khalf*16 + (col%16), vgpr j packs K = khalf*16+2j, +1
// One block per cluster. 128 threads.
// ---------------------------------------------------------------------------
__global__ __launch_bounds__(128) void prep_clusters(
    const float* __restrict__ centers, uint32_t* __restrict__ bws) {
    const int c   = blockIdx.x;        // cluster id 0..255
    const int tid = threadIdx.x;
    __shared__ float red[128];

    const float* row = centers + (size_t)c * DIM;
    float s = 0.f;
    for (int i = tid; i < DIM; i += 128) { float v = row[i]; s += v * v; }
    red[tid] = s;
    __syncthreads();
    for (int off = 64; off > 0; off >>= 1) {
        if (tid < off) red[tid] += red[tid + off];
        __syncthreads();
    }
    const float inv = 1.0f / fmaxf(sqrtf(red[0]), EPSN);

    const int ct = c >> 4;             // column tile
    const int cl = c & 15;             // column within tile
    for (int p = tid; p < DIM / 2; p += 128) {
        const int k     = 2 * p;
        const int kc    = k >> 5;
        const int rem   = k & 31;
        const int khalf = rem >> 4;
        const int j     = (rem & 15) >> 1;
        const int lane  = khalf * 16 + cl;
        bws[(((size_t)(kc * CTILES + ct) * 32 + lane) * 8) + j] =
            pack2h(row[k] * inv, row[k + 1] * inv);
    }
}

// ---------------------------------------------------------------------------
// Kernel 2: per-wave 16-row tile GEMM (raw f16 features x normalized f16
// clusters, f32 accum), argmax, outputs. 8 waves / block, 2048 blocks.
// __launch_bounds__(256, 2): force <=256 VGPRs/wave -> 4 waves/SIMD occupancy
// (the kernel is HBM-latency bound on the streaming feature read).
// ---------------------------------------------------------------------------
__global__ __launch_bounds__(256, 2) void kmeans_main(
    const float* __restrict__ feats, const float* __restrict__ weight,
    const int*   __restrict__ assign, const uint32_t* __restrict__ bmat,
    float* __restrict__ out, float* __restrict__ partials) {

    const int lane   = threadIdx.x & 31;
    const int wave   = threadIdx.x >> 5;
    const int tile   = blockIdx.x * 8 + wave;      // 16-row tile id
    const int row_in = lane & 15;                  // A layout: lane L and L+16 hold row L%16
    const size_t row_g = (size_t)tile * 16 + row_in;
    const float* frow = feats + row_g * DIM;
    const int koff = (lane < 16) ? 0 : 16;         // K sub-offset per lane half

    v8f acc[CTILES];
#pragma unroll
    for (int t = 0; t < CTILES; ++t) acc[t] = (v8f){0,0,0,0,0,0,0,0};

    float sumsq = 0.f;                             // this lane's half of |row|^2
    // Rolled loop: 16 independent WMMAs + clause-batched loads per iteration
    // give ILP; rolling keeps VGPRs under 256 for 4-waves/SIMD occupancy.
#pragma unroll 1
    for (int kc = 0; kc < KCHUNKS; ++kc) {
        const int k0 = kc * 32 + koff;
        // Features are streamed (each cacheline read once): non-temporal so
        // the resident 192KB B matrix is not evicted from WGP$/L2.
        const v4f l0 = __builtin_nontemporal_load((const v4f*)(frow + k0));
        const v4f l1 = __builtin_nontemporal_load((const v4f*)(frow + k0 + 4));
        const v4f h0 = __builtin_nontemporal_load((const v4f*)(frow + k0 + 8));
        const v4f h1 = __builtin_nontemporal_load((const v4f*)(frow + k0 + 12));
        sumsq += l0.x*l0.x + l0.y*l0.y + l0.z*l0.z + l0.w*l0.w
               + l1.x*l1.x + l1.y*l1.y + l1.z*l1.z + l1.w*l1.w
               + h0.x*h0.x + h0.y*h0.y + h0.z*h0.z + h0.w*h0.w
               + h1.x*h1.x + h1.y*h1.y + h1.z*h1.z + h1.w*h1.w;

        v16h a;
        a[0]=(_Float16)l0.x;  a[1]=(_Float16)l0.y;  a[2]=(_Float16)l0.z;  a[3]=(_Float16)l0.w;
        a[4]=(_Float16)l1.x;  a[5]=(_Float16)l1.y;  a[6]=(_Float16)l1.z;  a[7]=(_Float16)l1.w;
        a[8]=(_Float16)h0.x;  a[9]=(_Float16)h0.y;  a[10]=(_Float16)h0.z; a[11]=(_Float16)h0.w;
        a[12]=(_Float16)h1.x; a[13]=(_Float16)h1.y; a[14]=(_Float16)h1.z; a[15]=(_Float16)h1.w;

        const uint32_t* bbase = bmat + ((size_t)(kc * CTILES) * 32 + lane) * 8;
#pragma unroll
        for (int ct = 0; ct < CTILES; ++ct) {
            union { v8u u; v16h h; } bu;
            bu.u = *(const v8u*)(bbase + (size_t)ct * 32 * 8);
            acc[ct] = __builtin_amdgcn_wmma_f32_16x16x32_f16(
                false, a, false, bu.h, (short)0, acc[ct], false, false);
        }
    }

    // Full |row|^2: lane L and L^16 hold the two halves of row L%16.
    const float normsq = sumsq + __shfl_xor(sumsq, 16, 32);
    const float invn   = 1.0f / fmaxf(sqrtf(normsq), EPSN);   // inv-norm of row (lane&15)

    // Argmax per row. C layout: vgpr r, lanes 0-15 -> M=r, lanes 16-31 -> M=8+r,
    // column N = ct*16 + (lane&15). Reduce across the 16 lanes of each half.
    float bestv[8];
    int   besti[8];
    const int col0 = lane & 15;
#pragma unroll
    for (int r = 0; r < 8; ++r) {
        float bv = acc[0][r];
        int   bi = col0;
#pragma unroll
        for (int ct = 1; ct < CTILES; ++ct) {
            const float v = acc[ct][r];
            const int   c = ct * 16 + col0;
            if (v > bv) { bv = v; bi = c; }       // strict > keeps first index
        }
#pragma unroll
        for (int m = 1; m < 16; m <<= 1) {
            const float ov = __shfl_xor(bv, m, 32);
            const int   oi = __shfl_xor(bi, m, 32);
            if (ov > bv || (ov == bv && oi < bi)) { bv = ov; bi = oi; }
        }
        bestv[r] = bv; besti[r] = bi;
    }

    // Writers: lane r -> row tile*16+r (uses own invn), lane 16+r -> row tile*16+8+r
    // (needs inv-norm of row 8+r, held by lane 8+r).
    float lterm = 0.f;
#pragma unroll
    for (int r = 0; r < 8; ++r) {
        const float nhi = __shfl(invn, 8 + r, 32);
        if (lane == r) {
            const size_t m = (size_t)tile * 16 + r;
            const int lbl = besti[r];
            __builtin_nontemporal_store((float)lbl, out + m);
            __builtin_nontemporal_store((float)assign[lbl], out + NPTS + m);
            lterm += -bestv[r] * invn * weight[m];
        }
        if (lane == 16 + r) {
            const size_t m = (size_t)tile * 16 + 8 + r;
            const int lbl = besti[r];
            __builtin_nontemporal_store((float)lbl, out + m);
            __builtin_nontemporal_store((float)assign[lbl], out + NPTS + m);
            lterm += -bestv[r] * nhi * weight[m];
        }
    }

    // Deterministic loss partial: wave reduce -> block reduce -> partials[block]
#pragma unroll
    for (int m = 16; m > 0; m >>= 1) lterm += __shfl_xor(lterm, m, 32);
    __shared__ float wsum[8];
    if (lane == 0) wsum[wave] = lterm;
    __syncthreads();
    if (threadIdx.x == 0) {
        float s = 0.f;
#pragma unroll
        for (int i = 0; i < 8; ++i) s += wsum[i];
        partials[blockIdx.x] = s;
    }
}

// ---------------------------------------------------------------------------
// Kernel 3: final loss reduction (deterministic, single block)
// ---------------------------------------------------------------------------
__global__ __launch_bounds__(256) void final_reduce(
    const float* __restrict__ partials, int np, float* __restrict__ out) {
    __shared__ float red[256];
    float s = 0.f;
    for (int i = threadIdx.x; i < np; i += 256) s += partials[i];
    red[threadIdx.x] = s;
    __syncthreads();
    for (int off = 128; off > 0; off >>= 1) {
        if (threadIdx.x < off) red[threadIdx.x] += red[threadIdx.x + off];
        __syncthreads();
    }
    if (threadIdx.x == 0) out[2 * (size_t)NPTS] = red[0] / (float)NPTS;
}

extern "C" void kernel_launch(void* const* d_in, const int* in_sizes, int n_in,
                              void* d_out, int out_size, void* d_ws, size_t ws_size,
                              hipStream_t stream) {
    const float* feats   = (const float*)d_in[0];   // [4,256,256,384] f32
    const float* weight  = (const float*)d_in[1];   // [4,256,256] f32
    const float* centers = (const float*)d_in[2];   // [256,384] f32
    const int*   assign  = (const int*)  d_in[3];   // [256] i32

    uint32_t* bws      = (uint32_t*)d_ws;                              // 192KB B matrix
    float*    partials = (float*)((char*)d_ws + (size_t)B_DWORDS * 4); // 2048 floats

    prep_clusters<<<NCLASS, 128, 0, stream>>>(centers, bws);
    kmeans_main<<<NPTS / 128, 256, 0, stream>>>(feats, weight, assign, bws,
                                                (float*)d_out, partials);
    final_reduce<<<1, 256, 0, stream>>>(partials, NPTS / 128, (float*)d_out);
}
